// GATWithDimensionalityReduction_49108656062563
// MI455X (gfx1250) — compile-verified, matrix operand order checked by reading, not run on hardware
//
#include <hip/hip_runtime.h>
#include <hip/hip_bf16.h>
#include <math.h>

// ---------------------------------------------------------------------------
// GAT (4x GATConv, H=4 heads) + MLP head for gfx1250 (CDNA5, wave32, WMMA).
// - Weights converted f32->f16 and permuted into the CDNA5 WMMA B-register
//   layout ONCE (pack_weights), stored per-N-tile as contiguous slabs.
// - GEMM blocks DMA their K x 16 weight slab into LDS with the Tensor Data
//   Mover (tensor_load_to_lds + s_wait_tensorcnt), then run fully-unrolled
//   v_wmma_f32_16x16x32_f16 with A fed by global_load_b128 + v_cvt_pk_f16_f32.
// - Edge softmax/aggregation -> f32 atomics (working set fits 192MB L2).
// ---------------------------------------------------------------------------

#define HEADS 4
#define LRELU_SLOPE 0.2f

typedef __attribute__((ext_vector_type(16))) _Float16 v16h;
typedef __attribute__((ext_vector_type(8)))  float    v8f;
typedef __attribute__((ext_vector_type(4)))  unsigned v4u;
typedef __attribute__((ext_vector_type(8)))  int      v8i;
typedef __attribute__((ext_vector_type(4)))  int      v4i;

#if defined(__has_builtin)
#  if __has_builtin(__builtin_amdgcn_tensor_load_to_lds) && \
      __has_builtin(__builtin_amdgcn_s_wait_tensorcnt)
#    define USE_TDM 1
#  else
#    define USE_TDM 0
#  endif
#else
#  define USE_TDM 0
#endif

// ----- monotone float<->uint encoding for atomicMax on floats ---------------
__device__ __forceinline__ unsigned f32_ord(float f) {
    unsigned b = __float_as_uint(f);
    return b ^ ((b & 0x80000000u) ? 0xFFFFFFFFu : 0x80000000u);
}
__device__ __forceinline__ float ord_f32(unsigned k) {
    unsigned b = k ^ ((k & 0x80000000u) ? 0x80000000u : 0xFFFFFFFFu);
    return __uint_as_float(b);
}
#define NEG_INF_ORD 0x007FFFFFu   // f32_ord(-inf)

__device__ __forceinline__ void get_edge(const int* __restrict__ ei, int E,
                                         int eid, int& s, int& d) {
    if (eid < E) { s = ei[eid]; d = ei[E + eid]; }
    else         { s = eid - E; d = eid - E; }     // appended self-loop
}

__device__ __forceinline__ float elu1(float x) {
    return x > 0.0f ? x : (expf(x) - 1.0f);
}

// ---------------------------------------------------------------------------
// One-time weight pack: W[K x Nc] f32 -> per-N-tile f16 slabs in the CDNA5
// 16-bit B-matrix register layout. Consumer of element (k, n within tile):
//   kk = k % 32, hi = (kk >> 3) & 1, lane = n + 16*hi,
//   e = (kk & 7) + (kk & 16 ? 8 : 0)
// Slab for tile t lives at Wp + t*K*16, element index (k/32)*512 + lane*16 + e
// so each lane's 16 halves per k-step are 32 contiguous bytes.
// ---------------------------------------------------------------------------
__global__ void pack_weights(const float* __restrict__ W, _Float16* __restrict__ Wp,
                             int K, int Nc) {
    int i = blockIdx.x * blockDim.x + threadIdx.x;
    if (i >= K * Nc) return;
    int k = i / Nc, n = i % Nc;
    int tile = n >> 4, nn = n & 15;
    int kb = k >> 5, kk = k & 31;
    int lane = nn + (((kk >> 3) & 1) << 4);
    int e    = (kk & 7) + ((kk & 16) ? 8 : 0);
    Wp[(size_t)tile * K * 16 + (kb << 9) + (lane << 4) + e] = (_Float16)W[(size_t)k * Nc + n];
}

// ---------------------------------------------------------------------------
// WMMA GEMM: C[M,Nc] = A[M,K] * B[K,Nc], B pre-packed (see above).
// Preconditions at every call site: M % 16 == 0, Nc % 16 == 0, K in
// {32,64,128,256}. Block = 256 threads = 8 waves = 8 M-tiles x 1 N-tile.
// ---------------------------------------------------------------------------
template <int K>
__global__ void gemm_f16_wmma(const float* __restrict__ A,
                              const _Float16* __restrict__ Wp,
                              float* __restrict__ C,
                              int M, int Nc) {
    constexpr int KB   = K / 32;
    constexpr int SLAB = K * 16;              // halves; SLAB*2 bytes <= 8KB
    __shared__ __align__(32) _Float16 Bs[SLAB];

    int tilesN = Nc >> 4;
    int bn  = blockIdx.x % tilesN;
    int bm8 = blockIdx.x / tilesN;
    int n0  = bn << 4;
    const _Float16* src = Wp + (size_t)bn * SLAB;

#if USE_TDM
    // ---- TDM: one DMA of the pre-packed slab into LDS (wave 0 only) ----
    if (threadIdx.x < 32) {
        unsigned lds_off = (unsigned)(uintptr_t)(&Bs[0]);
        unsigned long long ga = (unsigned long long)(uintptr_t)src;
        v4u g0 = { 1u,                                   // count=1, user desc
                   lds_off,                              // D#.lds_addr
                   (unsigned)(ga & 0xFFFFFFFFu),         // global_addr[31:0]
                   (unsigned)(((ga >> 32) & 0x01FFFFFFu) // global_addr[56:32]
                              | (2u << 30)) };           // type=2 ("image")
        v8i g1 = { (int)0x10000,                         // data_size=1 -> 2B
                   (int)((SLAB & 0xFFFF) << 16),         // tensor_dim0[15:0]
                   (int)(((unsigned)SLAB >> 16) | (1u << 16)), // dim0 hi | tensor_dim1=1
                   (int)((unsigned)SLAB << 16),          // tile_dim0 = SLAB
                   0,                                    // tile_dim1=0, tile_dim2=0
                   (int)SLAB,                            // tensor_dim0_stride lo
                   0, 0 };
        v4i gz4 = { 0, 0, 0, 0 };
        v8i gz8 = { 0, 0, 0, 0, 0, 0, 0, 0 };
        __builtin_amdgcn_tensor_load_to_lds(g0, g1, gz4, gz4, gz8, 0);
        __builtin_amdgcn_s_wait_tensorcnt(0);
    }
    __syncthreads();
#else
    // ---- fallback: cooperative 16B copies ----
    for (int i = threadIdx.x; i < SLAB / 8; i += 256)
        reinterpret_cast<ulonglong2*>(Bs)[i] =
            reinterpret_cast<const ulonglong2*>(src)[i];
    __syncthreads();
#endif

    int wave = threadIdx.x >> 5;
    int lane = threadIdx.x & 31;
    int tm   = bm8 * 8 + wave;
    if (tm * 16 >= M) return;                 // wave-uniform: EXEC stays full

    int m0    = tm << 4;
    int mrow  = m0 + (lane & 15);
    int kbase = (lane < 16) ? 0 : 8;
    const float* arow = A + (size_t)mrow * K;

    v8f acc = {};
#pragma unroll
    for (int kb = 0; kb < KB; ++kb) {
        int k0 = kb << 5;
        const float4* ap0 = reinterpret_cast<const float4*>(arow + k0 + kbase);
        const float4* ap1 = reinterpret_cast<const float4*>(arow + k0 + kbase + 16);
        float4 a0 = ap0[0], a1 = ap0[1];
        float4 a2 = ap1[0], a3 = ap1[1];

        v16h a;
        a[0]  = (_Float16)a0.x; a[1]  = (_Float16)a0.y;
        a[2]  = (_Float16)a0.z; a[3]  = (_Float16)a0.w;
        a[4]  = (_Float16)a1.x; a[5]  = (_Float16)a1.y;
        a[6]  = (_Float16)a1.z; a[7]  = (_Float16)a1.w;
        a[8]  = (_Float16)a2.x; a[9]  = (_Float16)a2.y;
        a[10] = (_Float16)a2.z; a[11] = (_Float16)a2.w;
        a[12] = (_Float16)a3.x; a[13] = (_Float16)a3.y;
        a[14] = (_Float16)a3.z; a[15] = (_Float16)a3.w;

        v16h b = *reinterpret_cast<const v16h*>(&Bs[(kb << 9) + (lane << 4)]);

        acc = __builtin_amdgcn_wmma_f32_16x16x32_f16(
            /*neg_a=*/false, a, /*neg_b=*/false, b,
            /*c_mod=*/(short)0, acc, /*reuse_a=*/false, /*reuse_b=*/false);
    }

    int mbase = m0 + ((lane < 16) ? 0 : 8);
    int nn    = n0 + (lane & 15);
#pragma unroll
    for (int r = 0; r < 8; ++r)
        C[(size_t)(mbase + r) * Nc + nn] = acc[r];
}

// ----- per-(node,head) attention projections alpha_s / alpha_d --------------
__global__ void alpha_kernel(const float* __restrict__ h,
                             const float* __restrict__ a_src,
                             const float* __restrict__ a_dst,
                             float* __restrict__ as_n,
                             float* __restrict__ ad_n,
                             int N, int C) {
    int idx = blockIdx.x * blockDim.x + threadIdx.x;
    if (idx >= N * HEADS) return;
    int hd = idx & (HEADS - 1);
    int i  = idx >> 2;
    const float* hp = h + ((size_t)i * HEADS + hd) * C;
    const float* ws = a_src + (size_t)hd * C;
    const float* wd = a_dst + (size_t)hd * C;
    float ss = 0.0f, sd = 0.0f;
    for (int c = 0; c < C; ++c) {
        float v = hp[c];
        ss = fmaf(v, ws[c], ss);
        sd = fmaf(v, wd[c], sd);
    }
    as_n[idx] = ss;
    ad_n[idx] = sd;
}

// ----- buffer init ----------------------------------------------------------
__global__ void init_seg(unsigned* __restrict__ emax, float* __restrict__ denom, int n) {
    int i = blockIdx.x * blockDim.x + threadIdx.x;
    if (i < n) { emax[i] = NEG_INF_ORD; denom[i] = 0.0f; }
}
__global__ void zero_f32(float* __restrict__ p, size_t n) {
    size_t i = (size_t)blockIdx.x * blockDim.x + threadIdx.x;
    if (i < n) p[i] = 0.0f;
}

// ----- segment max over destination (leaky-relu'd logits) -------------------
__global__ void edge_max(const int* __restrict__ ei,
                         const float* __restrict__ as_n,
                         const float* __restrict__ ad_n,
                         unsigned* __restrict__ emax,
                         int E, int E2) {
    int idx = blockIdx.x * blockDim.x + threadIdx.x;
    if (idx >= E2 * HEADS) return;
    int hd = idx & (HEADS - 1);
    int eid = idx >> 2;
    int s, d; get_edge(ei, E, eid, s, d);
    float e = as_n[s * HEADS + hd] + ad_n[d * HEADS + hd];
    e = e > 0.0f ? e : LRELU_SLOPE * e;
    atomicMax(&emax[d * HEADS + hd], f32_ord(e));
}

// ----- exp(e - emax[dst]) and denominator -----------------------------------
__global__ void edge_exp(const int* __restrict__ ei,
                         const float* __restrict__ as_n,
                         const float* __restrict__ ad_n,
                         const unsigned* __restrict__ emax,
                         float* __restrict__ denom,
                         float* __restrict__ ee,
                         int E, int E2) {
    int idx = blockIdx.x * blockDim.x + threadIdx.x;
    if (idx >= E2 * HEADS) return;
    int hd = idx & (HEADS - 1);
    int eid = idx >> 2;
    int s, d; get_edge(ei, E, eid, s, d);
    float e = as_n[s * HEADS + hd] + ad_n[d * HEADS + hd];
    e = e > 0.0f ? e : LRELU_SLOPE * e;
    float v = expf(e - ord_f32(emax[d * HEADS + hd]));
    ee[idx] = v;
    atomicAdd(&denom[d * HEADS + hd], v);
}

// ----- message scatter: out[dst] += h[src] * alpha --------------------------
__global__ void edge_scatter(const int* __restrict__ ei,
                             const float* __restrict__ h,
                             const float* __restrict__ ee,
                             const float* __restrict__ denom,
                             float* __restrict__ out,
                             int E, int E2, int C) {
    int idx = blockIdx.x * blockDim.x + threadIdx.x;
    if (idx >= E2 * HEADS) return;
    int hd = idx & (HEADS - 1);
    int eid = idx >> 2;
    int s, d; get_edge(ei, E, eid, s, d);
    float alpha = ee[idx] / denom[d * HEADS + hd];
    const float4* hp = reinterpret_cast<const float4*>(h + ((size_t)s * HEADS + hd) * C);
    float*        op = out + ((size_t)d * HEADS + hd) * C;
    for (int c4 = 0; c4 < (C >> 2); ++c4) {
        float4 hv = hp[c4];
        atomicAdd(&op[c4 * 4 + 0], hv.x * alpha);
        atomicAdd(&op[c4 * 4 + 1], hv.y * alpha);
        atomicAdd(&op[c4 * 4 + 2], hv.z * alpha);
        atomicAdd(&op[c4 * 4 + 3], hv.w * alpha);
    }
}

// ----- bias + ELU epilogue --------------------------------------------------
__global__ void bias_elu(float* __restrict__ x, const float* __restrict__ b,
                         int N, int D) {
    size_t i = (size_t)blockIdx.x * blockDim.x + threadIdx.x;
    if (i >= (size_t)N * D) return;
    int c = (int)(i % D);
    x[i] = elu1(x[i] + b[c]);
}

// ----- final [N,64] @ [64,1] + bias -> d_out --------------------------------
__global__ void final_dot(const float* __restrict__ h,
                          const float* __restrict__ w,
                          const float* __restrict__ b,
                          float* __restrict__ out, int N, int K) {
    int i = blockIdx.x * blockDim.x + threadIdx.x;
    if (i >= N) return;
    const float* hp = h + (size_t)i * K;
    float s = 0.0f;
    for (int k = 0; k < K; ++k) s = fmaf(hp[k], w[k], s);
    out[i] = s + b[0];
}

// ---------------------------------------------------------------------------
// Host-side orchestration
// ---------------------------------------------------------------------------
static inline unsigned cdiv(long a, long b) { return (unsigned)((a + b - 1) / b); }

static void launch_pack(const float* W, _Float16* Wp, int K, int Nc, hipStream_t s) {
    pack_weights<<<cdiv((long)K * Nc, 256), 256, 0, s>>>(W, Wp, K, Nc);
}

static void launch_gemm(const float* A, const _Float16* Wp, float* C,
                        int M, int K, int Nc, hipStream_t s) {
    int tilesM = M >> 4;
    int tilesN = Nc >> 4;
    unsigned blocks = cdiv(tilesM, 8) * tilesN;
    switch (K) {
        case 32:  gemm_f16_wmma<32> <<<blocks, 256, 0, s>>>(A, Wp, C, M, Nc); break;
        case 64:  gemm_f16_wmma<64> <<<blocks, 256, 0, s>>>(A, Wp, C, M, Nc); break;
        case 128: gemm_f16_wmma<128><<<blocks, 256, 0, s>>>(A, Wp, C, M, Nc); break;
        case 256: gemm_f16_wmma<256><<<blocks, 256, 0, s>>>(A, Wp, C, M, Nc); break;
    }
}

static void gat_layer(const float* xin, int K, int C,
                      const _Float16* Wp, const float* asw, const float* adw,
                      const float* bias,
                      float* hbuf, float* outbuf,
                      float* as_n, float* ad_n, unsigned* emax, float* denom,
                      float* ee, const int* ei, int N, int E, hipStream_t s) {
    int D  = HEADS * C;
    int E2 = E + N;
    launch_gemm(xin, Wp, hbuf, N, K, D, s);
    alpha_kernel<<<cdiv((long)N * HEADS, 256), 256, 0, s>>>(hbuf, asw, adw, as_n, ad_n, N, C);
    init_seg    <<<cdiv((long)N * HEADS, 256), 256, 0, s>>>(emax, denom, N * HEADS);
    zero_f32    <<<cdiv((long)N * D, 256),     256, 0, s>>>(outbuf, (size_t)N * D);
    edge_max    <<<cdiv((long)E2 * HEADS, 256), 256, 0, s>>>(ei, as_n, ad_n, emax, E, E2);
    edge_exp    <<<cdiv((long)E2 * HEADS, 256), 256, 0, s>>>(ei, as_n, ad_n, emax, denom, ee, E, E2);
    edge_scatter<<<cdiv((long)E2 * HEADS, 256), 256, 0, s>>>(ei, hbuf, ee, denom, outbuf, E, E2, C);
    bias_elu    <<<cdiv((long)N * D, 256),     256, 0, s>>>(outbuf, bias, N, D);
}

extern "C" void kernel_launch(void* const* d_in, const int* in_sizes, int n_in,
                              void* d_out, int out_size, void* d_ws, size_t ws_size,
                              hipStream_t stream) {
    const int N = in_sizes[0] / 128;     // 50000
    const int E = in_sizes[1] / 2;       // 800000

    const float* x   = (const float*)d_in[0];
    const int*   ei  = (const int*)  d_in[1];
    // d_in[2] = edge_attr (unused, as in reference)
    const float* W1 = (const float*)d_in[3],  *as1 = (const float*)d_in[4],
               * ad1 = (const float*)d_in[5], *b1  = (const float*)d_in[6];
    const float* W2 = (const float*)d_in[7],  *as2 = (const float*)d_in[8],
               * ad2 = (const float*)d_in[9], *b2  = (const float*)d_in[10];
    const float* W3 = (const float*)d_in[11], *as3 = (const float*)d_in[12],
               * ad3 = (const float*)d_in[13],*b3  = (const float*)d_in[14];
    const float* W4 = (const float*)d_in[15], *as4 = (const float*)d_in[16],
               * ad4 = (const float*)d_in[17],*b4  = (const float*)d_in[18];
    const float* Wr  = (const float*)d_in[19], *br  = (const float*)d_in[20];
    const float* Wf1 = (const float*)d_in[21], *bf1 = (const float*)d_in[22];
    const float* Wf2 = (const float*)d_in[23], *bf2 = (const float*)d_in[24];

    // --- carve workspace ---
    char* p = (char*)d_ws;
    auto carve = [&](size_t bytes) -> void* {
        void* r = (void*)p;
        p += (bytes + 255) & ~(size_t)255;
        return r;
    };
    float*    hbuf  = (float*)   carve((size_t)N * 256 * 4);
    float*    bufA  = (float*)   carve((size_t)N * 256 * 4);
    float*    bufB  = (float*)   carve((size_t)N * 256 * 4);
    float*    as_n  = (float*)   carve((size_t)N * HEADS * 4);
    float*    ad_n  = (float*)   carve((size_t)N * HEADS * 4);
    unsigned* emax  = (unsigned*)carve((size_t)N * HEADS * 4);
    float*    denom = (float*)   carve((size_t)N * HEADS * 4);
    float*    ee    = (float*)   carve((size_t)(E + N) * HEADS * 4);
    _Float16* wp1   = (_Float16*)carve((size_t)128 * 32  * 2);
    _Float16* wp2   = (_Float16*)carve((size_t)32  * 64  * 2);
    _Float16* wp3   = (_Float16*)carve((size_t)64  * 128 * 2);
    _Float16* wp4   = (_Float16*)carve((size_t)128 * 256 * 2);
    _Float16* wpr   = (_Float16*)carve((size_t)256 * 32  * 2);
    _Float16* wpf1  = (_Float16*)carve((size_t)32  * 64  * 2);
    (void)ws_size; (void)n_in; (void)out_size;

    // --- one-time weight pack (f32 -> f16, WMMA B-register layout) ---
    launch_pack(W1,  wp1,  128, 32,  stream);
    launch_pack(W2,  wp2,  32,  64,  stream);
    launch_pack(W3,  wp3,  64,  128, stream);
    launch_pack(W4,  wp4,  128, 256, stream);
    launch_pack(Wr,  wpr,  256, 32,  stream);
    launch_pack(Wf1, wpf1, 32,  64,  stream);

    // --- 4 GAT layers (ELU fused into each layer epilogue) ---
    gat_layer(x,    128,  8, wp1, as1, ad1, b1, hbuf, bufA, as_n, ad_n, emax, denom, ee, ei, N, E, stream);
    gat_layer(bufA,  32, 16, wp2, as2, ad2, b2, hbuf, bufB, as_n, ad_n, emax, denom, ee, ei, N, E, stream);
    gat_layer(bufB,  64, 32, wp3, as3, ad3, b3, hbuf, bufA, as_n, ad_n, emax, denom, ee, ei, N, E, stream);
    gat_layer(bufA, 128, 64, wp4, as4, ad4, b4, hbuf, bufB, as_n, ad_n, emax, denom, ee, ei, N, E, stream);

    // --- MLP head ---
    launch_gemm(bufB, wpr, hbuf, N, 256, 32, stream);
    bias_elu<<<cdiv((long)N * 32, 256), 256, 0, stream>>>(hbuf, br, N, 32);
    launch_gemm(hbuf, wpf1, bufA, N, 32, 64, stream);
    bias_elu<<<cdiv((long)N * 64, 256), 256, 0, stream>>>(bufA, bf1, N, 64);
    final_dot<<<cdiv(N, 256), 256, 0, stream>>>(bufA, Wf2, bf2, (float*)d_out, N, 64);
}
